// Non_Local_62663572849138
// MI455X (gfx1250) — matrix operand order repaired and checked by reading, not compiled, not run on hardware
//
#include <hip/hip_runtime.h>
#include <hip/hip_bf16.h>

// Problem constants (reference: B=8, C=256, CP=64, H=W=64, N=4096)
#define BATCH 8
#define C_IN  256
#define CPRJ  64
#define N_SP  4096

typedef _Float16 v16h __attribute__((ext_vector_type(16)));
typedef _Float16 v8h  __attribute__((ext_vector_type(8)));
typedef float    v8f  __attribute__((ext_vector_type(8)));
typedef unsigned int u32x4 __attribute__((ext_vector_type(4)));
typedef int          i32x4 __attribute__((ext_vector_type(4)));
typedef int          i32x8 __attribute__((ext_vector_type(8)));

// ---------- fragment helpers ----------
__device__ __forceinline__ v16h ld_frag2(const _Float16* lo_p, const _Float16* hi_p) {
    v8h lo = *(const v8h*)lo_p;
    v8h hi = *(const v8h*)hi_p;
    v16h r;
#pragma unroll
    for (int i = 0; i < 8; ++i) { r[i] = lo[i]; r[i + 8] = hi[i]; }
    return r;
}
// B-fragment column: p = column n (lane&15) at k-row 0; gather K rows (stride ld) in table order
__device__ __forceinline__ v16h ld_frag16(const _Float16* p, int ld) {
    int kh8 = ((threadIdx.x & 31) >> 4) << 3;
    v16h r;
#pragma unroll
    for (int i = 0; i < 8; ++i) { r[i] = p[(kh8 + i) * ld]; r[i + 8] = p[(16 + kh8 + i) * ld]; }
    return r;
}
__device__ __forceinline__ v8f wmma_f16(v16h a, v16h b, v8f c) {
    return __builtin_amdgcn_wmma_f32_16x16x32_f16(false, a, false, b, (short)0, c, false, false);
}

// ---------- Tensor Data Mover: 2D tile (fp16) global -> LDS ----------
// LDS padding by the DMA engine: 4 DWORDs inserted after every 32 DWORDs,
// i.e. a 64-half (128B) tile row lands on a 144B (72-half) LDS stride.
__device__ __forceinline__ void tdm_load_2d(unsigned lds_off, const void* gptr,
                                            unsigned dim0, unsigned dim1,
                                            unsigned tile0, unsigned tile1,
                                            unsigned stride0) {
    unsigned long long ga = (unsigned long long)gptr;
    u32x4 g0;
    g0.x = 1u;                                     // count=1 (valid descriptor)
    g0.y = lds_off;                                // LDS byte address
    g0.z = (unsigned)ga;                           // global addr [31:0]
    g0.w = (unsigned)(ga >> 32) | 0x80000000u;     // global addr [56:32], type=2
    i32x8 g1;
    g1[0] = (int)((1u << 16) | (1u << 20) | (4u << 22) | (3u << 25));
    //            data_size=2B | pad_en  | pad_int=32dw | pad_amt=4dw
    g1[1] = (int)((dim0 & 0xffffu) << 16);                       // tensor_dim0 lo16
    g1[2] = (int)((dim0 >> 16) | ((dim1 & 0xffffu) << 16));      // dim0 hi / dim1 lo
    g1[3] = (int)((dim1 >> 16) | (tile0 << 16));                 // dim1 hi / tile_dim0
    g1[4] = (int)tile1;                                          // tile_dim1 (tile_dim2=0)
    g1[5] = (int)stride0;                                        // tensor_dim0_stride lo32
    g1[6] = 0;
    g1[7] = 0;
    i32x4 gz = {0, 0, 0, 0};
#if __clang_major__ >= 23
    i32x8 gz8 = {0, 0, 0, 0, 0, 0, 0, 0};
    __builtin_amdgcn_tensor_load_to_lds(g0, g1, gz, gz, gz8, 0);
#else
    __builtin_amdgcn_tensor_load_to_lds(g0, g1, gz, gz, 0);
#endif
}

// ---------- phase 0: Wo^T (fp16, const2 folded) ----------
__global__ __launch_bounds__(256) void wot_kernel(const float* __restrict__ Wo,
                                                  _Float16* __restrict__ wot) {
    const float const2 = 1.41421f / 8.0f;  // sqrt2 / sqrt(CP)
    int idx = blockIdx.x * 256 + threadIdx.x;   // 0 .. 64*256-1
    int c  = idx >> 8;
    int co = idx & 255;
    wot[c * C_IN + co] = (_Float16)(Wo[co * CPRJ + c] * const2);
}

// ---------- phase 1: fused QKV projection ----------
// q[b][n][cp], v[b][n][cp] (row-major over channels), k[b][cp][n] (channel-major)
__global__ __launch_bounds__(256) void qkv_kernel(
    const float* __restrict__ x,
    const float* __restrict__ Wq, const float* __restrict__ bq,
    const float* __restrict__ Wk, const float* __restrict__ bk,
    const float* __restrict__ Wv, const float* __restrict__ bv,
    _Float16* __restrict__ qh, _Float16* __restrict__ kh, _Float16* __restrict__ vh) {
    const float const1 = 1.41421f / 16.0f;  // sqrt2 / sqrt(C)
    __shared__ _Float16 xs[C_IN][72];       // 64 cols + pad (avoid bank conflicts)

    int b  = blockIdx.x >> 6;
    int n0 = (blockIdx.x & 63) << 6;
    int t  = threadIdx.x;                   // == channel for staging

    {   // stage const1 * x tile (256 channels x 64 columns) as fp16
        const float* xr = x + ((size_t)(b * C_IN + t)) * N_SP + n0;
#pragma unroll 8
        for (int j = 0; j < 64; ++j) xs[t][j] = (_Float16)(xr[j] * const1);
    }
    __syncthreads();

    int o    = t & 63;          // output channel 0..63
    int col0 = (t >> 6) << 4;   // 16-column slice
    float aq[16], ak[16], av[16];
#pragma unroll
    for (int j = 0; j < 16; ++j) { aq[j] = 0.f; ak[j] = 0.f; av[j] = 0.f; }

    const float* wqr = Wq + (size_t)o * C_IN;
    const float* wkr = Wk + (size_t)o * C_IN;
    const float* wvr = Wv + (size_t)o * C_IN;
    for (int c = 0; c < C_IN; ++c) {
        float wq = wqr[c], wk = wkr[c], wv = wvr[c];
#pragma unroll
        for (int j = 0; j < 16; ++j) {
            float xv = (float)xs[c][col0 + j];
            aq[j] = fmaf(wq, xv, aq[j]);
            ak[j] = fmaf(wk, xv, ak[j]);
            av[j] = fmaf(wv, xv, av[j]);
        }
    }
    float bqo = bq[o], bko = bk[o], bvo = bv[o];
#pragma unroll
    for (int j = 0; j < 16; ++j) {
        int n = n0 + col0 + j;
        qh[((size_t)(b * N_SP + n)) * CPRJ + o] = (_Float16)(aq[j] + bqo);
        vh[((size_t)(b * N_SP + n)) * CPRJ + o] = (_Float16)(av[j] + bvo);
        kh[((size_t)(b * CPRJ + o)) * N_SP + n] = (_Float16)(ak[j] + bko);
    }
}

// ---------- phase 2: fused flash attention + output projection + residual ----------
// one wave (32 lanes) per 16-query tile; 4 waves per block share TDM-staged K/V tiles
__global__ __launch_bounds__(128) void attn_kernel(
    const _Float16* __restrict__ qh, const _Float16* __restrict__ kh,
    const _Float16* __restrict__ vh, const _Float16* __restrict__ wot,
    const float* __restrict__ bo, const float* __restrict__ gamma,
    const float* __restrict__ x, float* __restrict__ out) {
    __shared__ _Float16 ldsK[2][CPRJ][72];  // K tile: [channel][key], TDM-padded rows
    __shared__ _Float16 ldsV[2][64][72];    // V tile: [key][channel], TDM-padded rows
    __shared__ _Float16 pT[4][16][64];      // per-wave P staging (C/D -> A layout)
    __shared__ _Float16 oT[4][16][64];      // per-wave O staging

    int wave = threadIdx.x >> 5;
    int lane = threadIdx.x & 31;
    int gw   = (blockIdx.x << 2) + wave;      // 0..2047 (all 4 waves: same batch)
    int b    = gw >> 8;                       // 256 query tiles per batch
    int q0   = (gw & 255) << 4;
    int row  = lane & 15;                     // A-row / C-D column index
    int hi16 = lane >> 4;
    int khalf = hi16 << 3;                    // A-fragment K split: 0 or 8

    const _Float16* kbase = kh + (size_t)b * CPRJ * N_SP;
    const _Float16* vbase = vh + (size_t)b * N_SP * CPRJ;

    // Q fragments (16 queries x 64 channels, K-chunks of 32)
    const _Float16* qrow = qh + ((size_t)(b * N_SP + q0 + row)) * CPRJ;
    v16h aQ0 = ld_frag2(qrow + khalf,      qrow + khalf + 16);
    v16h aQ1 = ld_frag2(qrow + 32 + khalf, qrow + 48 + khalf);

    // all-ones B fragment: row-sum of P through the matrix pipe (flash "ones" trick)
    v16h vOnes;
#pragma unroll
    for (int i = 0; i < 16; ++i) vOnes[i] = (_Float16)1.0f;

    float m[8];
    v8f O0 = {}, O1 = {}, O2 = {}, O3 = {}, L = {};
#pragma unroll
    for (int i = 0; i < 8; ++i) m[i] = -1e30f;
    const v8f zc = {};

    // prologue: TDM-stage first K/V block into buffer 0
    if (wave == 0) {
        tdm_load_2d((unsigned)(size_t)(const void*)&ldsK[0][0][0], kbase,
                    N_SP, CPRJ, 64, CPRJ, N_SP);          // K: rows=channels, stride N
        tdm_load_2d((unsigned)(size_t)(const void*)&ldsV[0][0][0], vbase,
                    CPRJ, N_SP, CPRJ, 64, CPRJ);          // V: rows=keys, stride CP
        __builtin_amdgcn_s_wait_tensorcnt((short)0);
    }
    __syncthreads();

    for (int j0 = 0; j0 < N_SP; j0 += 64) {
        int buf = (j0 >> 6) & 1;
        if (wave == 0 && (j0 + 64) < N_SP) {   // TDM prefetch next block into other buffer
            tdm_load_2d((unsigned)(size_t)(const void*)&ldsK[buf ^ 1][0][0],
                        kbase + (j0 + 64), N_SP, CPRJ, 64, CPRJ, N_SP);
            tdm_load_2d((unsigned)(size_t)(const void*)&ldsV[buf ^ 1][0][0],
                        vbase + (size_t)(j0 + 64) * CPRJ, CPRJ, N_SP, CPRJ, 64, CPRJ);
        }
        const _Float16* kT = &ldsK[buf][0][0];
        const _Float16* vT = &ldsV[buf][0][0];
        const _Float16* k0 = kT + row;                        // channels 0..31
        const _Float16* k1 = kT + (size_t)32 * 72 + row;      // channels 32..63

        // 4 score tiles of 16 keys, K = 64 channels in 2 WMMA k-steps
        v8f s0 = wmma_f16(aQ0, ld_frag16(k0, 72),      zc); s0 = wmma_f16(aQ1, ld_frag16(k1, 72),      s0);
        v8f s1 = wmma_f16(aQ0, ld_frag16(k0 + 16, 72), zc); s1 = wmma_f16(aQ1, ld_frag16(k1 + 16, 72), s1);
        v8f s2 = wmma_f16(aQ0, ld_frag16(k0 + 32, 72), zc); s2 = wmma_f16(aQ1, ld_frag16(k1 + 32, 72), s2);
        v8f s3 = wmma_f16(aQ0, ld_frag16(k0 + 48, 72), zc); s3 = wmma_f16(aQ1, ld_frag16(k1 + 48, 72), s3);

        // online softmax: per C/D row (element i <-> row i + 8*hi16, col = lane&15)
        float mn_[8];
#pragma unroll
        for (int i = 0; i < 8; ++i) {
            float rmax = fmaxf(fmaxf(s0[i], s1[i]), fmaxf(s2[i], s3[i]));
            rmax = fmaxf(rmax, __shfl_xor(rmax, 1, 32));
            rmax = fmaxf(rmax, __shfl_xor(rmax, 2, 32));
            rmax = fmaxf(rmax, __shfl_xor(rmax, 4, 32));
            rmax = fmaxf(rmax, __shfl_xor(rmax, 8, 32));
            float mn = fmaxf(m[i], rmax);
            float sc = __expf(m[i] - mn);
            m[i] = mn;
            mn_[i] = mn;
            L[i]  *= sc;
            O0[i] *= sc; O1[i] *= sc; O2[i] *= sc; O3[i] *= sc;
        }
        // exp + stage P to LDS (C/D layout -> A layout)
#pragma unroll
        for (int i = 0; i < 8; ++i) {
            int M = i + (hi16 << 3);
            pT[wave][M][row]      = (_Float16)__expf(s0[i] - mn_[i]);
            pT[wave][M][row + 16] = (_Float16)__expf(s1[i] - mn_[i]);
            pT[wave][M][row + 32] = (_Float16)__expf(s2[i] - mn_[i]);
            pT[wave][M][row + 48] = (_Float16)__expf(s3[i] - mn_[i]);
        }
        asm volatile("s_wait_dscnt 0x0" ::: "memory");
        v16h aP0 = ld_frag2(&pT[wave][row][khalf],      &pT[wave][row][khalf + 16]);
        v16h aP1 = ld_frag2(&pT[wave][row][32 + khalf], &pT[wave][row][48 + khalf]);

        // softmax denominator via matrix pipe: L += P x 1
        L = wmma_f16(aP0, vOnes, L);
        L = wmma_f16(aP1, vOnes, L);

        // O += P x V (V B-fragments: lane = channel column, elements gather key rows)
        const _Float16* vrow0 = vT + row;                        // keys j0..j0+31
        const _Float16* vrow1 = vT + (size_t)32 * 72 + row;      // keys j0+32..j0+63
        O0 = wmma_f16(aP0, ld_frag16(vrow0, 72),      O0);
        O0 = wmma_f16(aP1, ld_frag16(vrow1, 72),      O0);
        O1 = wmma_f16(aP0, ld_frag16(vrow0 + 16, 72), O1);
        O1 = wmma_f16(aP1, ld_frag16(vrow1 + 16, 72), O1);
        O2 = wmma_f16(aP0, ld_frag16(vrow0 + 32, 72), O2);
        O2 = wmma_f16(aP1, ld_frag16(vrow1 + 32, 72), O2);
        O3 = wmma_f16(aP0, ld_frag16(vrow0 + 48, 72), O3);
        O3 = wmma_f16(aP1, ld_frag16(vrow1 + 48, 72), O3);

        // next buffer must be resident before anyone proceeds past the barrier
        if (wave == 0) __builtin_amdgcn_s_wait_tensorcnt((short)0);
        __syncthreads();
    }

    // normalize and stage O (fp16) for the projection A-fragments
#pragma unroll
    for (int i = 0; i < 8; ++i) {
        float inv = 1.0f / L[i];
        O0[i] *= inv; O1[i] *= inv; O2[i] *= inv; O3[i] *= inv;
    }
#pragma unroll
    for (int i = 0; i < 8; ++i) {
        int M = i + (hi16 << 3);
        oT[wave][M][row]      = (_Float16)O0[i];
        oT[wave][M][row + 16] = (_Float16)O1[i];
        oT[wave][M][row + 32] = (_Float16)O2[i];
        oT[wave][M][row + 48] = (_Float16)O3[i];
    }
    asm volatile("s_wait_dscnt 0x0" ::: "memory");
    v16h aO0 = ld_frag2(&oT[wave][row][khalf],      &oT[wave][row][khalf + 16]);
    v16h aO1 = ld_frag2(&oT[wave][row][32 + khalf], &oT[wave][row][48 + khalf]);

    // output projection: out[b][co][n] = gamma*(Wo*const2*sa + bo) + x
    float g = gamma[0];
#pragma unroll
    for (int coc = 0; coc < 16; ++coc) {
        const _Float16* w0 = wot + coc * 16 + row;                        // c = 0..31
        const _Float16* w1 = wot + (size_t)32 * C_IN + coc * 16 + row;    // c = 32..63
        v8f acc = wmma_f16(aO0, ld_frag16(w0, C_IN), zc);
        acc     = wmma_f16(aO1, ld_frag16(w1, C_IN), acc);

        int co = (coc << 4) + row;
        float boc = bo[co];
        size_t base = ((size_t)(b * C_IN + co)) * N_SP + q0 + (hi16 << 3);
#pragma unroll
        for (int i = 0; i < 8; ++i) {
            out[base + i] = g * (acc[i] + boc) + x[base + i];
        }
    }
}

extern "C" void kernel_launch(void* const* d_in, const int* in_sizes, int n_in,
                              void* d_out, int out_size, void* d_ws, size_t ws_size,
                              hipStream_t stream) {
    const float* x     = (const float*)d_in[0];
    const float* Wq    = (const float*)d_in[1];
    const float* bq    = (const float*)d_in[2];
    const float* Wk    = (const float*)d_in[3];
    const float* bk    = (const float*)d_in[4];
    const float* Wv    = (const float*)d_in[5];
    const float* bv    = (const float*)d_in[6];
    const float* Wo    = (const float*)d_in[7];
    const float* bo    = (const float*)d_in[8];
    const float* gamma = (const float*)d_in[9];
    float* out = (float*)d_out;

    char* ws = (char*)d_ws;
    const size_t qkv_bytes = (size_t)BATCH * N_SP * CPRJ * sizeof(_Float16); // 4 MB each
    _Float16* qh  = (_Float16*)(ws);
    _Float16* kh  = (_Float16*)(ws + qkv_bytes);
    _Float16* vh  = (_Float16*)(ws + 2 * qkv_bytes);
    _Float16* wot = (_Float16*)(ws + 3 * qkv_bytes);

    wot_kernel<<<64, 256, 0, stream>>>(Wo, wot);
    qkv_kernel<<<BATCH * (N_SP / 64), 256, 0, stream>>>(x, Wq, bq, Wk, bk, Wv, bv, qh, kh, vh);
    attn_kernel<<<(BATCH * (N_SP / 16)) / 4, 128, 0, stream>>>(qh, kh, vh, wot, bo, gamma, x, out);
}
